// HLGHeadLayers_83897891160259
// MI455X (gfx1250) — matrix operand order repaired and checked by reading, not compiled, not run on hardware
//
#include <hip/hip_runtime.h>
#include <hip/hip_bf16.h>
#include <hip/hip_fp16.h>

typedef __attribute__((ext_vector_type(16))) _Float16 v16h;
typedef __attribute__((ext_vector_type(8)))  float    v8f;
typedef unsigned int u32;
typedef int v4i __attribute__((__vector_size__(16)));

#define WMMA_F16(a,b,c) __builtin_amdgcn_wmma_f32_16x16x32_f16(false,(a),false,(b),(short)0,(c),false,false)

#if __has_builtin(__builtin_amdgcn_global_load_async_to_lds_b128)
#define HAVE_ASYNC_LDS 1
#define AS1 __attribute__((address_space(1)))
#define AS3 __attribute__((address_space(3)))
#else
#define HAVE_ASYNC_LDS 0
#endif

// copy 16B global -> LDS (async DMA if available, sync load/store otherwise)
__device__ __forceinline__ void cp16(const _Float16* g, _Float16* l) {
#if HAVE_ASYNC_LDS
  __builtin_amdgcn_global_load_async_to_lds_b128(
      (AS1 v4i*)(unsigned long long)(const void*)g,
      (AS3 v4i*)(unsigned int)(unsigned long long)(void*)l,
      0, 0);
#else
  *(int4*)l = *(const int4*)g;
#endif
}
__device__ __forceinline__ void wait_async() {
#if HAVE_ASYNC_LDS
#if __has_builtin(__builtin_amdgcn_s_wait_asynccnt)
  __builtin_amdgcn_s_wait_asynccnt(0);
#else
  asm volatile("s_wait_asynccnt 0x0" ::: "memory");
#endif
#endif
}

// ---------------- WMMA fragment loaders (16x16x32 f16, contiguous k order in A and B) ----------------
// A (16x32, MxK): lane L: m=L%16, half=L/16. VGPR j: K {2j,2j+1}+16*half
__device__ __forceinline__ v16h load_a_frag(const _Float16* base, int ld, int lane) {
  int m = lane & 15, half = lane >> 4;
  const _Float16* r = base + (size_t)m * ld + 16 * half;
  v16h a; u32* ap = (u32*)&a;
  ap[0] = *(const u32*)(r + 0);  ap[1] = *(const u32*)(r + 2);
  ap[2] = *(const u32*)(r + 4);  ap[3] = *(const u32*)(r + 6);
  ap[4] = *(const u32*)(r + 8);  ap[5] = *(const u32*)(r + 10);
  ap[6] = *(const u32*)(r + 12); ap[7] = *(const u32*)(r + 14);
  return a;
}
// B (32x16, KxN) from Bt row-major [N,K]: lane L: col n=L%16, VGPR j: K {2j,2j+1}+16*(L/16)
__device__ __forceinline__ v16h load_b_frag(const _Float16* base, int ld, int lane) {
  int n = lane & 15, half = lane >> 4;
  const _Float16* r = base + (size_t)n * ld + 16 * half;
  v16h b; u32* bp = (u32*)&b;
#pragma unroll
  for (int j = 0; j < 8; ++j) bp[j] = *(const u32*)(r + 2 * j);
  return b;
}
// C/D frag: VGPR r -> row (r + 8*(lane/16)), col lane%16.

// ---------------- LayerNorm: f32 in -> f16 out ----------------
__global__ __launch_bounds__(256) void ln_kernel(const float* __restrict__ in,
                                                 const float* __restrict__ g,
                                                 const float* __restrict__ b,
                                                 _Float16* __restrict__ out, int C) {
  int row = blockIdx.x;
  const float* x = in + (size_t)row * C;
  __shared__ float red[256];
  float s = 0.f, s2 = 0.f;
  for (int i = threadIdx.x; i < C; i += 256) { float v = x[i]; s += v; s2 += v * v; }
  red[threadIdx.x] = s; __syncthreads();
  for (int st = 128; st > 0; st >>= 1) { if (threadIdx.x < st) red[threadIdx.x] += red[threadIdx.x + st]; __syncthreads(); }
  float mean = red[0] / (float)C; __syncthreads();
  red[threadIdx.x] = s2; __syncthreads();
  for (int st = 128; st > 0; st >>= 1) { if (threadIdx.x < st) red[threadIdx.x] += red[threadIdx.x + st]; __syncthreads(); }
  float var = red[0] / (float)C - mean * mean;
  float inv = rsqrtf(var + 1e-5f);
  for (int i = threadIdx.x; i < C; i += 256)
    out[(size_t)row * C + i] = (_Float16)((x[i] - mean) * inv * g[i] + b[i]);
}

// ---------------- weight transpose+convert: f32 [K,N] -> f16 [N,K] ----------------
__global__ void wtrans(const float* __restrict__ in, _Float16* __restrict__ out,
                       int K, int N) {
  size_t idx = (size_t)blockIdx.x * blockDim.x + threadIdx.x;
  size_t total = (size_t)K * N;
  if (idx >= total) return;
  int k = (int)(idx % K); int n = (int)(idx / K);
  out[idx] = (_Float16)in[(size_t)k * N + n];
}

// ---------------- V transpose: kv16 [B*N, 2C] (v in cols C+h*d+kd) -> vT [B,h,d,N] ----
__global__ void vtrans(const _Float16* __restrict__ kv16, _Float16* __restrict__ vT,
                       int N, int C, int heads, int d) {
  size_t idx = (size_t)blockIdx.x * blockDim.x + threadIdx.x;
  int n = (int)(idx % N); size_t r = idx / N;
  int kd = (int)(r % d); size_t r2 = r / d;
  int h = (int)(r2 % heads); int b = (int)(r2 / heads);
  vT[idx] = kv16[((size_t)b * N + n) * (2 * C) + C + h * d + kd];
}

// ---------------- cooperative WMMA GEMM: workgroup = 32x256 tile, 8 waves of 16x64 ------
// A,B staged through double-buffered LDS with async-to-lds DMA overlap.
// mode 0: out16 = acc ; mode 1: out16 = gelu(acc + bias) ; mode 2: out32 = resid + acc + bias
__global__ __launch_bounds__(256) void gemm_wmma(
    const _Float16* __restrict__ A, const _Float16* __restrict__ Bt,
    const float* __restrict__ bias, const float* __restrict__ resid,
    _Float16* __restrict__ out16, float* __restrict__ out32,
    int M, int N, int K, int mode) {
  __shared__ __align__(16) _Float16 sA[2][32][72];   // 32 rows x 64 k (+8 pad)
  __shared__ __align__(16) _Float16 sB[2][256][72];  // 256 cols x 64 k (+8 pad)
  const int t = threadIdx.x;
  const int lane = t & 31, wave = t >> 5;
  const int rh = wave >> 2;      // row half: 0/1
  const int cq = wave & 3;       // col quarter: 0..3
  const int row0 = blockIdx.x * 32;
  const int col0 = blockIdx.y * 256;
  if (row0 >= M || col0 >= N) return;

  auto fill = [&](int s, int k0) {
    int ar = t >> 3, ac = (t & 7) * 8;
    cp16(A + (size_t)(row0 + ar) * K + k0 + ac, &sA[s][ar][ac]);
    const _Float16* bg = Bt + (size_t)(col0 + t) * K + k0;
#pragma unroll
    for (int j = 0; j < 8; ++j) cp16(bg + j * 8, &sB[s][t][j * 8]);
  };

  v8f c0 = {}, c1 = {}, c2 = {}, c3 = {};
  int s = 0;
  fill(0, 0);
  wait_async();
  __syncthreads();
  for (int k0 = 0; k0 < K; k0 += 64) {
    if (k0 + 64 < K) fill(s ^ 1, k0 + 64);   // overlap next-stage DMA with compute
#pragma unroll
    for (int ks = 0; ks < 64; ks += 32) {
      v16h a   = load_a_frag(&sA[s][rh * 16][ks], 72, lane);
      v16h bb0 = load_b_frag(&sB[s][cq * 64 +  0][ks], 72, lane);
      v16h bb1 = load_b_frag(&sB[s][cq * 64 + 16][ks], 72, lane);
      v16h bb2 = load_b_frag(&sB[s][cq * 64 + 32][ks], 72, lane);
      v16h bb3 = load_b_frag(&sB[s][cq * 64 + 48][ks], 72, lane);
      c0 = WMMA_F16(a, bb0, c0);
      c1 = WMMA_F16(a, bb1, c1);
      c2 = WMMA_F16(a, bb2, c2);
      c3 = WMMA_F16(a, bb3, c3);
    }
    wait_async();
    __syncthreads();
    s ^= 1;
  }

  const int half = lane >> 4, lm = lane & 15;
  v8f cacc[4] = {c0, c1, c2, c3};
#pragma unroll
  for (int q = 0; q < 4; ++q) {
#pragma unroll
    for (int r = 0; r < 8; ++r) {
      int gm = row0 + rh * 16 + r + 8 * half;
      int gn = col0 + cq * 64 + q * 16 + lm;
      float v = cacc[q][r];
      if (mode == 1) {
        v += bias[gn];
        v = 0.5f * v * (1.f + erff(v * 0.70710678118f));
        out16[(size_t)gm * N + gn] = (_Float16)v;
      } else if (mode == 2) {
        v += bias[gn];
        out32[(size_t)gm * N + gn] = resid[(size_t)gm * N + gn] + v;
      } else {
        out16[(size_t)gm * N + gn] = (_Float16)v;
      }
    }
  }
}

// ---------------- dynamic position bias MLP: [(2H-1)(2W-1), 2] -> [P, heads] ----------------
__global__ __launch_bounds__(64) void posmlp(
    const float* __restrict__ ppw, const float* __restrict__ ppb,
    const float* __restrict__ g1, const float* __restrict__ b1,
    const float* __restrict__ w1, const float* __restrict__ bb1,
    const float* __restrict__ g2, const float* __restrict__ b2,
    const float* __restrict__ w2, const float* __restrict__ bb2,
    const float* __restrict__ g3, const float* __restrict__ b3,
    const float* __restrict__ w3, const float* __restrict__ bb3,
    float* __restrict__ posT, int H, int W, int pd, int heads) {
  int p = blockIdx.x, t = threadIdx.x;
  int W2 = 2 * W - 1;
  float bh = (float)(p / W2 + 1 - H);
  float bw = (float)(p % W2 + 1 - W);
  __shared__ float sv[64];
  __shared__ float red[64];
  float x = bh * ppw[t] + bw * ppw[pd + t] + ppb[t];

  auto reduce = [&](float val) -> float {
    red[t] = val; __syncthreads();
    for (int st = 32; st > 0; st >>= 1) { if (t < st) red[t] += red[t + st]; __syncthreads(); }
    float r = red[0]; __syncthreads(); return r;
  };
  auto ln_relu_store = [&](float xv, const float* gg, const float* bb) {
    float mu = reduce(xv) / (float)pd;
    float dv = xv - mu;
    float var = reduce(dv * dv) / (float)pd;
    float y = dv * rsqrtf(var + 1e-5f) * gg[t] + bb[t];
    y = fmaxf(y, 0.f);
    sv[t] = y; __syncthreads();
  };

  ln_relu_store(x, g1, b1);
  float acc = bb1[t];
  for (int k = 0; k < 64; ++k) acc += sv[k] * w1[k * pd + t];
  __syncthreads(); x = acc;

  ln_relu_store(x, g2, b2);
  acc = bb2[t];
  for (int k = 0; k < 64; ++k) acc += sv[k] * w2[k * pd + t];
  __syncthreads(); x = acc;

  ln_relu_store(x, g3, b3);
  if (t < heads) {
    float o = bb3[t];
    for (int k = 0; k < 64; ++k) o += sv[k] * w3[k * heads + t];
    posT[(size_t)p * heads + t] = o;
  }
}

// ---------------- fused attention: 8 waves cooperate on one (b, h, 16-row strip) ----------------
// score pass: column tiles strided over waves; softmax: 16 threads/row; PV: K-split over
// waves with partial accumulators reduced through LDS (reusing the dead score buffer).
#define AN 1024
__global__ __launch_bounds__(256) void attn_kernel(
    const _Float16* __restrict__ q16,  // [B*N, C]
    const _Float16* __restrict__ kv16, // [B*N, 2C] (k in first C cols)
    const _Float16* __restrict__ vT16, // [B, h, d, N]
    const float* __restrict__ posT,    // [(2H-1)(2W-1), heads]
    _Float16* __restrict__ o16,        // [B*N, C]
    int N, int C, int heads, int d, int H, int W, float scale) {
  __shared__ __align__(16) float    sS[16 * AN];   // scores, later reused as o-partials
  __shared__ __align__(16) _Float16 sP[16 * AN];   // softmax numerator (f16)
  __shared__ float sMax[16], sSum[16], sRed[16][17];
  int b = blockIdx.x / heads, h = blockIdx.x % heads;
  int row0 = blockIdx.y * 16;
  int t = threadIdx.x;
  int lane = t & 31, w = t >> 5;           // 8 waves
  int lm = lane & 15, half = lane >> 4;

  const _Float16* qbase = q16 + ((size_t)b * N) * C + h * d;
  v16h aq0 = load_a_frag(qbase + (size_t)row0 * C + 0,  C, lane);
  v16h aq1 = load_a_frag(qbase + (size_t)row0 * C + 32, C, lane);

  // scores = q @ k^T -> LDS (column tiles strided over the 8 waves)
  const _Float16* kbase = kv16 + ((size_t)b * N) * (2 * C) + h * d;
  for (int ct = w; ct < N / 16; ct += 8) {
    v16h b0 = load_b_frag(kbase + (size_t)(ct * 16) * (2 * C) + 0,  2 * C, lane);
    v16h b1 = load_b_frag(kbase + (size_t)(ct * 16) * (2 * C) + 32, 2 * C, lane);
    v8f c = {};
    c = WMMA_F16(aq0, b0, c);
    c = WMMA_F16(aq1, b1, c);
#pragma unroll
    for (int r = 0; r < 8; ++r) sS[(r + 8 * half) * AN + ct * 16 + lm] = c[r];
  }
  __syncthreads();

  // softmax: 16 threads per row, 64 elements each; fold scale + analytic rel-pos bias
  int row = t & 15, seg = t >> 4;          // seg: 0..15
  int jj0 = seg * (N / 16), jj1 = jj0 + N / 16;
  int gm = row0 + row;
  int hi = gm / W, wi = gm % W;
  int W2 = 2 * W - 1;
  float* srow = sS + (size_t)row * AN;
  _Float16* prow = sP + (size_t)row * AN;
  float mx = -3.4e38f;
  for (int j = jj0; j < jj1; ++j) {
    int hj = j / W, wj = j % W;
    int rel = (hi - hj + H - 1) * W2 + (wi - wj + W - 1);
    float sv = srow[j] * scale + posT[(size_t)rel * heads + h];
    srow[j] = sv;
    mx = fmaxf(mx, sv);
  }
  sRed[row][seg] = mx; __syncthreads();
  if (t < 16) {
    float m = -3.4e38f;
    for (int k = 0; k < 16; ++k) m = fmaxf(m, sRed[t][k]);
    sMax[t] = m;
  }
  __syncthreads();
  float rmax = sMax[row];
  float ls = 0.f;
  for (int j = jj0; j < jj1; ++j) {
    float e = expf(srow[j] - rmax);
    ls += e;
    prow[j] = (_Float16)e;                 // unnormalized numerator
  }
  sRed[row][seg] = ls; __syncthreads();
  if (t < 16) {
    float sum = 0.f;
    for (int k = 0; k < 16; ++k) sum += sRed[t][k];
    sSum[t] = sum;
  }
  __syncthreads();

  // PV: each wave reduces 4 of the 32 K-blocks; partials to LDS, then final reduce
  v8f o0 = {}, o1 = {}, o2 = {}, o3 = {};
  const _Float16* vb = vT16 + ((size_t)(b * heads + h)) * d * N;
  for (int ct = w * 4; ct < w * 4 + 4; ++ct) {
    v16h ap  = load_a_frag(sP + ct * 32, AN, lane);
    v16h vb0 = load_b_frag(vb + (size_t)( 0) * N + ct * 32, N, lane);
    v16h vb1 = load_b_frag(vb + (size_t)(16) * N + ct * 32, N, lane);
    v16h vb2 = load_b_frag(vb + (size_t)(32) * N + ct * 32, N, lane);
    v16h vb3 = load_b_frag(vb + (size_t)(48) * N + ct * 32, N, lane);
    o0 = WMMA_F16(ap, vb0, o0);
    o1 = WMMA_F16(ap, vb1, o1);
    o2 = WMMA_F16(ap, vb2, o2);
    o3 = WMMA_F16(ap, vb3, o3);
  }
  float* sO = sS;                          // reuse score buffer: [8 waves][16 rows][64 d]
  v8f oo[4] = {o0, o1, o2, o3};
#pragma unroll
  for (int q = 0; q < 4; ++q) {
#pragma unroll
    for (int r = 0; r < 8; ++r) {
      int m = r + 8 * half;
      sO[((size_t)w * 16 + m) * 64 + q * 16 + lm] = oo[q][r];
    }
  }
  __syncthreads();
  for (int i = t; i < 16 * 64; i += 256) {
    int m = i >> 6, dc = i & 63;
    float acc = 0.f;
    for (int ww = 0; ww < 8; ++ww) acc += sO[((size_t)ww * 16 + m) * 64 + dc];
    o16[((size_t)b * N + row0 + m) * C + h * d + dc] = (_Float16)(acc / sSum[m]);
  }
}

// ---------------- host orchestration ----------------
extern "C" void kernel_launch(void* const* d_in, const int* in_sizes, int n_in,
                              void* d_out, int out_size, void* d_ws, size_t ws_size,
                              hipStream_t stream) {
  const float* x       = (const float*)d_in[0];
  const float* n1g     = (const float*)d_in[1];
  const float* n1b     = (const float*)d_in[2];
  const float* q_w     = (const float*)d_in[3];
  const float* kv_w    = (const float*)d_in[4];
  const float* proj_w  = (const float*)d_in[5];
  const float* proj_b  = (const float*)d_in[6];
  const float* ppw     = (const float*)d_in[7];
  const float* ppb     = (const float*)d_in[8];
  const float* g1      = (const float*)d_in[9];
  const float* b1      = (const float*)d_in[10];
  const float* w1      = (const float*)d_in[11];
  const float* bb1     = (const float*)d_in[12];
  const float* g2      = (const float*)d_in[13];
  const float* b2      = (const float*)d_in[14];
  const float* w2      = (const float*)d_in[15];
  const float* bb2     = (const float*)d_in[16];
  const float* g3      = (const float*)d_in[17];
  const float* b3      = (const float*)d_in[18];
  const float* w3      = (const float*)d_in[19];
  const float* bb3     = (const float*)d_in[20];
  const float* n2g     = (const float*)d_in[21];
  const float* n2b     = (const float*)d_in[22];
  const float* fc1_w   = (const float*)d_in[23];
  const float* fc1_b   = (const float*)d_in[24];
  const float* fc2_w   = (const float*)d_in[25];
  const float* fc2_b   = (const float*)d_in[26];

  const int C     = in_sizes[1];            // 1024
  const int hid   = in_sizes[23] / C;       // 4096
  const int heads = in_sizes[20];           // 16
  const int pd    = in_sizes[8];            // 64
  const int H = 32, W = 32;                 // fixed problem geometry (H*W == N)
  const int N  = H * W;                     // 1024
  const int M  = in_sizes[0] / C;           // B*N = 4096
  const int Bsz = M / N;                    // 4
  const int d  = C / heads;                 // 64
  const int P  = (2 * H - 1) * (2 * W - 1); // 3969
  const float scale = rsqrtf((float)d);

  size_t off = 0;
  auto carve = [&](size_t bytes) -> void* {
    void* p = (char*)d_ws + off;
    off += (bytes + 255) & ~(size_t)255;
    return p;
  };
  _Float16* xn16   = (_Float16*)carve((size_t)M * C * 2);
  _Float16* qWt    = (_Float16*)carve((size_t)C * C * 2);
  _Float16* kvWt   = (_Float16*)carve((size_t)2 * C * C * 2);
  _Float16* projWt = (_Float16*)carve((size_t)C * C * 2);
  _Float16* fc1Wt  = (_Float16*)carve((size_t)hid * C * 2);
  _Float16* fc2Wt  = (_Float16*)carve((size_t)C * hid * 2);
  _Float16* q16    = (_Float16*)carve((size_t)M * C * 2);
  _Float16* kv16   = (_Float16*)carve((size_t)M * 2 * C * 2);
  _Float16* vT16   = (_Float16*)carve((size_t)M * C * 2);
  float*    posT   = (float*)   carve((size_t)P * heads * 4);
  _Float16* ao16   = (_Float16*)carve((size_t)M * C * 2);
  float*    x1     = (float*)   carve((size_t)M * C * 4);
  _Float16* h16    = (_Float16*)carve((size_t)M * C * 2);
  _Float16* g16    = (_Float16*)carve((size_t)M * hid * 2);

  // 1) norm1 -> f16
  ln_kernel<<<M, 256, 0, stream>>>(x, n1g, n1b, xn16, C);

  // 2) weight transpose/convert
  auto gw = [](size_t tot) { return dim3((unsigned)((tot + 255) / 256)); };
  wtrans<<<gw((size_t)C * C),   256, 0, stream>>>(q_w,    qWt,    C,   C);
  wtrans<<<gw((size_t)C * 2*C), 256, 0, stream>>>(kv_w,   kvWt,   C,   2 * C);
  wtrans<<<gw((size_t)C * C),   256, 0, stream>>>(proj_w, projWt, C,   C);
  wtrans<<<gw((size_t)C * hid), 256, 0, stream>>>(fc1_w,  fc1Wt,  C,   hid);
  wtrans<<<gw((size_t)hid * C), 256, 0, stream>>>(fc2_w,  fc2Wt,  hid, C);

  // 3) q / kv projections (WMMA, async LDS-staged)
  gemm_wmma<<<dim3(M / 32, C / 256),     256, 0, stream>>>(xn16, qWt,  nullptr, nullptr, q16,  nullptr, M, C,     C, 0);
  gemm_wmma<<<dim3(M / 32, 2 * C / 256), 256, 0, stream>>>(xn16, kvWt, nullptr, nullptr, kv16, nullptr, M, 2 * C, C, 0);

  // 4) v transpose for PV WMMA; pos-bias MLP
  vtrans<<<gw((size_t)M * C), 256, 0, stream>>>(kv16, vT16, N, C, heads, d);
  posmlp<<<P, pd, 0, stream>>>(ppw, ppb, g1, b1, w1, bb1, g2, b2, w2, bb2,
                               g3, b3, w3, bb3, posT, H, W, pd, heads);

  // 5) fused attention (8 waves per 16-row strip)
  attn_kernel<<<dim3(Bsz * heads, N / 16), 256, 0, stream>>>(q16, kv16, vT16, posT, ao16,
                                                             N, C, heads, d, H, W, scale);

  // 6) proj + residual -> x1 (f32)
  gemm_wmma<<<dim3(M / 32, C / 256), 256, 0, stream>>>(ao16, projWt, proj_b, x, nullptr, x1, M, C, C, 2);

  // 7) norm2 -> f16
  ln_kernel<<<M, 256, 0, stream>>>(x1, n2g, n2b, h16, C);

  // 8) fc1 + gelu -> f16
  gemm_wmma<<<dim3(M / 32, hid / 256), 256, 0, stream>>>(h16, fc1Wt, fc1_b, nullptr, g16, nullptr, M, hid, C, 1);

  // 9) fc2 + residual -> d_out (f32)
  gemm_wmma<<<dim3(M / 32, C / 256), 256, 0, stream>>>(g16, fc2Wt, fc2_b, x1, nullptr, (float*)d_out, M, C, hid, 2);
}
